// ContrastiveLoss_40673340293967
// MI455X (gfx1250) — matrix operand order repaired and checked by reading, not compiled, not run on hardware
//
#include <hip/hip_runtime.h>
#include <hip/hip_bf16.h>
#include <math.h>

#define EPS 1e-8f
#define LAM_SM 9.0f
#define LAM_LSE 6.0f
#define MARGIN 0.2f

typedef __attribute__((ext_vector_type(16))) _Float16 v16h;
typedef __attribute__((ext_vector_type(8)))  float    v8f;

union FragH { v16h v; uint4 q[2]; };

__device__ __forceinline__ float wave_sum(float x) {
    for (int o = 16; o; o >>= 1) x += __shfl_xor(x, o, 32);
    return x;
}

// ---------------------------------------------------------------------------
// Kernel 1: fp32 -> fp16 conversion (im padded 36->48 rows), caption norms.
// Blocks 0..63: image b. Blocks 64..127: caption b-64.
// ---------------------------------------------------------------------------
__global__ __launch_bounds__(256) void prep_kernel(
    const float* __restrict__ im, const float* __restrict__ s,
    _Float16* __restrict__ imh, _Float16* __restrict__ sh,
    float* __restrict__ capn)
{
    const int b = blockIdx.x, tid = threadIdx.x;
    if (b < 64) {
        for (int idx = tid; idx < 48 * 512; idx += 256) {
            int r = idx >> 9, d = idx & 511;
            float v = (r < 36) ? im[((size_t)b * 36 + r) * 512 + d] : 0.0f;
            imh[(size_t)b * 48 * 512 + idx] = (_Float16)v;
        }
    } else {
        const int c = b - 64;
        for (int idx = tid; idx < 32 * 512; idx += 256)
            sh[(size_t)c * 32 * 512 + idx] = (_Float16)s[(size_t)c * 32 * 512 + idx];
        const int wave = tid >> 5, lane = tid & 31;
        for (int l = wave * 4; l < wave * 4 + 4; ++l) {
            float ss = 0.0f;
            for (int d = lane; d < 512; d += 32) {
                float v = s[((size_t)c * 32 + l) * 512 + d];
                ss += v * v;
            }
            ss = wave_sum(ss);
            if (lane == 0) capn[c * 32 + l] = sqrtf(ss);
        }
    }
}

// ---------------------------------------------------------------------------
// Kernel 2: per-image Gram matrices G_i[r][r'] = im[i,r,:].im[i,r',:], fp32.
// ---------------------------------------------------------------------------
__global__ __launch_bounds__(256) void gram_kernel(
    const float* __restrict__ im, float* __restrict__ G)
{
    const int i = blockIdx.x, tid = threadIdx.x;
    const float* base = im + (size_t)i * 36 * 512;
    for (int e = tid; e < 36 * 36; e += 256) {
        int r = e / 36, r2 = e % 36;
        const float4* p = (const float4*)(base + r  * 512);
        const float4* q = (const float4*)(base + r2 * 512);
        float acc = 0.0f;
        for (int k = 0; k < 128; ++k) {
            float4 a = p[k], bq = q[k];
            acc += a.x * bq.x + a.y * bq.y + a.z * bq.z + a.w * bq.w;
        }
        G[(size_t)i * 1296 + e] = acc;
    }
}

// ---------------------------------------------------------------------------
// Kernel 3: one block per (image i, caption c) pair.
//   waves 0..5 : attn = im_h[i] (48x512) @ s_h[c]^T (512x32) via WMMA f16->f32
//   wave  6    : w_num = f_s[c].w_im[i]
//   wave  7    : ||f_s[c]||, ||w_im[i]||
// then LDS postprocessing -> scores[i*64+c]
// ---------------------------------------------------------------------------
__global__ __launch_bounds__(256) void pair_kernel(
    const _Float16* __restrict__ imh, const _Float16* __restrict__ sh,
    const float* __restrict__ G, const float* __restrict__ capn,
    const float* __restrict__ fs, const float* __restrict__ wim,
    const int* __restrict__ sl, float* __restrict__ scores)
{
    const int i = blockIdx.x, c = blockIdx.y;
    const int tid = threadIdx.x;
    const int wave = tid >> 5, lane = tid & 31;
    const int len = sl[c];

    __shared__ float attn[48 * 32];   // raw attn (rows >=36 are zero padding)
    __shared__ float smb[36 * 32];    // normalized a, then softmax values
    __shared__ float Gl[36 * 36];
    __shared__ float mxl[32], invl[32], numl[32];
    __shared__ float part[8][32];
    __shared__ float s_wnum, s_nf, s_nw;

    if (wave < 6) {
        // --- WMMA tile (mt,nt): 16x16 output, K=512 in steps of 32 ---
        const int mt = wave >> 1, nt = wave & 1;
        const int arow = mt * 16 + (lane & 15);
        const int kba  = (lane & 16) ? 8 : 0;     // A-fragment lane layout
        const int bcol = nt * 16 + (lane & 15);
        const int kbb  = (lane & 16) ? 8 : 0;     // B-fragment lane layout
        const _Float16* Ap = imh + ((size_t)i * 48 + arow) * 512;
        const _Float16* Bp = sh  + ((size_t)c * 32 + bcol) * 512;
        v8f acc = {};
        for (int k0 = 0; k0 < 512; k0 += 32) {
            FragH a, b;
            a.q[0] = *(const uint4*)(Ap + k0 + kba);        // K = k0+kba .. +7
            a.q[1] = *(const uint4*)(Ap + k0 + 16 + kba);   // K = k0+16+kba ..
            b.q[0] = *(const uint4*)(Bp + k0 + kbb);        // K = k0+kbb .. +7
            b.q[1] = *(const uint4*)(Bp + k0 + 16 + kbb);   // K = k0+16+kbb ..
            acc = __builtin_amdgcn_wmma_f32_16x16x32_f16(
                false, a.v, false, b.v, (short)0, acc, false, false);
        }
        const int nc = lane & 15;
        const int mb = (lane & 16) ? 8 : 0;
        #pragma unroll
        for (int v = 0; v < 8; ++v)
            attn[(mt * 16 + mb + v) * 32 + nt * 16 + nc] = acc[v];
    } else if (wave == 6) {
        float d = 0.0f;
        for (int k = lane; k < 512; k += 32) d += fs[c * 512 + k] * wim[i * 512 + k];
        d = wave_sum(d);
        if (lane == 0) s_wnum = d;
    } else {
        float a = 0.0f, b = 0.0f;
        for (int k = lane; k < 512; k += 32) {
            float x = fs[c * 512 + k];  a += x * x;
            float y = wim[i * 512 + k]; b += y * y;
        }
        a = wave_sum(a); b = wave_sum(b);
        if (lane == 0) { s_nf = sqrtf(a); s_nw = sqrtf(b); }
    }
    __syncthreads();

    // Stage Gram matrix into LDS (needed in step 3)
    for (int idx = tid; idx < 36 * 36; idx += 256) Gl[idx] = G[(size_t)i * 1296 + idx];

    // Step 1: a = leaky_relu(attn)*mask, l2-normalized over l (per region r)
    for (int r = wave; r < 36; r += 8) {
        float v = attn[r * 32 + lane];
        float a = (v > 0.0f) ? v : 0.1f * v;
        a = (lane < len) ? a : 0.0f;
        float ss = wave_sum(a * a);
        smb[r * 32 + lane] = a / (sqrtf(ss) + EPS);
    }
    __syncthreads();

    // Step 2a: softmax stats over r (per word l)
    if (tid < 32) {
        const int l = tid;
        float mx = -1e30f;
        for (int r = 0; r < 36; ++r) mx = fmaxf(mx, smb[r * 32 + l]);
        mx *= LAM_SM;
        float se = 0.0f;
        for (int r = 0; r < 36; ++r) se += expf(LAM_SM * smb[r * 32 + l] - mx);
        mxl[l] = mx; invl[l] = 1.0f / se; numl[l] = 0.0f;
    }
    __syncthreads();

    // Step 2b: materialize sm, accumulate num[l] = sum_r sm*attn_raw
    for (int idx = tid; idx < 36 * 32; idx += 256) {
        int r = idx >> 5, l = idx & 31;
        float e = expf(LAM_SM * smb[idx] - mxl[l]) * invl[l];
        smb[idx] = e;
        atomicAdd(&numl[l], e * attn[r * 32 + l]);
    }
    __syncthreads();

    // Step 3: ||wei[l]||^2 = sm[:,l]^T G sm[:,l]  (partials over 8 r'-chunks)
    {
        const int l = tid & 31, ch = tid >> 5;
        float acc = 0.0f;
        for (int j = 0; j < 5; ++j) {
            int rp = ch * 5 + j;
            if (rp < 36) {
                float srl = smb[rp * 32 + l];
                float inner = 0.0f;
                for (int r = 0; r < 36; ++r) inner += Gl[rp * 36 + r] * smb[r * 32 + l];
                acc += srl * inner;
            }
        }
        part[ch][l] = acc;
    }
    __syncthreads();

    // Step 4: row_sim, masked LSEs, final score
    if (tid < 32) {
        const int l = tid;
        float wn2 = 0.0f;
        for (int ch = 0; ch < 8; ++ch) wn2 += part[ch][l];
        float wn = sqrtf(fmaxf(wn2, 0.0f));
        float denom = fmaxf(capn[c * 32 + l] * wn, EPS);
        float rs = numl[l] / denom;
        float maxattn = invl[l];           // max of softmax = 1/sum(exp(shifted))
        bool m = (l < len);
        float e1 = m ? expf(LAM_LSE * rs)      : 0.0f;
        float e2 = m ? expf(LAM_LSE * maxattn) : 0.0f;
        e1 = wave_sum(e1); e2 = wave_sum(e2);
        if (l == 0) {
            float row_lse  = logf(e1) / LAM_LSE;
            float attn_lse = logf(e2) / LAM_LSE;
            float whole = s_wnum / fmaxf(s_nf * s_nw, EPS);
            scores[(size_t)i * 64 + c] = row_lse + 0.5f * attn_lse + 0.5f * whole;
        }
    }
}

// ---------------------------------------------------------------------------
// Kernel 4: hinge loss reduction over the 64x64 score matrix.
// ---------------------------------------------------------------------------
__global__ __launch_bounds__(64) void loss_kernel(
    const float* __restrict__ S, float* __restrict__ out)
{
    __shared__ float red[64];
    const int j = threadIdx.x;                 // row i=j and column c=j
    const float dj = S[j * 64 + j];
    float m1 = 0.0f, m2 = 0.0f;
    for (int k = 0; k < 64; ++k) {
        if (k != j) {
            float cs = MARGIN + S[j * 64 + k] - dj;  // cost_s row j (diag[i])
            m1 = fmaxf(m1, fmaxf(cs, 0.0f));
            float ci = MARGIN + S[k * 64 + j] - dj;  // cost_im col j (diag[c])
            m2 = fmaxf(m2, fmaxf(ci, 0.0f));
        }
    }
    red[j] = m1 + m2;
    __syncthreads();
    if (j == 0) {
        float s = 0.0f;
        for (int k = 0; k < 64; ++k) s += red[k];
        out[0] = s;
    }
}

// ---------------------------------------------------------------------------
extern "C" void kernel_launch(void* const* d_in, const int* in_sizes, int n_in,
                              void* d_out, int out_size, void* d_ws, size_t ws_size,
                              hipStream_t stream) {
    const float* im   = (const float*)d_in[0];   // (64,36,512)
    const float* w_im = (const float*)d_in[1];   // (64,512)
    const float* s    = (const float*)d_in[2];   // (64,32,512)
    const int*   s_l  = (const int*)  d_in[3];   // (64,)
    const float* f_s  = (const float*)d_in[4];   // (64,512)
    float* out = (float*)d_out;

    char* ws = (char*)d_ws;
    const size_t OFF_IMH  = 0;                        // 64*48*512*2 = 3,145,728
    const size_t OFF_SH   = OFF_IMH + 3145728;        // 64*32*512*2 = 2,097,152
    const size_t OFF_G    = OFF_SH  + 2097152;        // 64*1296*4   =   331,776
    const size_t OFF_CAPN = OFF_G   + 331776;         // 64*32*4     =     8,192
    const size_t OFF_SC   = OFF_CAPN + 8192;          // 64*64*4     =    16,384

    _Float16* imh  = (_Float16*)(ws + OFF_IMH);
    _Float16* sh   = (_Float16*)(ws + OFF_SH);
    float*    G    = (float*)   (ws + OFF_G);
    float*    capn = (float*)   (ws + OFF_CAPN);
    float*    sc   = (float*)   (ws + OFF_SC);

    prep_kernel<<<128, 256, 0, stream>>>(im, s, imh, sh, capn);
    gram_kernel<<<64, 256, 0, stream>>>(im, G);
    pair_kernel<<<dim3(64, 64), 256, 0, stream>>>(imh, sh, G, capn, f_s, w_im, s_l, sc);
    loss_kernel<<<1, 64, 0, stream>>>(sc, out);

    (void)in_sizes; (void)n_in; (void)out_size; (void)ws_size;
}
